// Dynamic_Critic_48009144434976
// MI455X (gfx1250) — matrix operand order repaired and checked by reading, not compiled, not run on hardware
//
#include <hip/hip_runtime.h>

#define B_   65536
#define T_   5
#define OBS_ 128
#define ACT_ 32
#define H_   64
#define BT_  (B_*T_)

typedef __attribute__((ext_vector_type(16))) _Float16 v16h;
typedef __attribute__((ext_vector_type(8)))  _Float16 v8h;
typedef __attribute__((ext_vector_type(4)))  _Float16 v4h;
typedef __attribute__((ext_vector_type(2)))  _Float16 v2h;
typedef __attribute__((ext_vector_type(8)))  float    v8f;

__device__ __forceinline__ v8f wmma16(v16h a, v16h b, v8f c){
  // D = A(16x32 f16) * B(32x16 f16) + C(f32)
  return __builtin_amdgcn_wmma_f32_16x16x32_f16(false, a, false, b, (short)0, c, false, false);
}

// A fragment (16x32 f16): one contiguous 16-half run per lane (k = 16*hi + s) -> 2x b128 loads.
// `row` = pointer to this lane's matrix row (M = lane&15), hi = lane>>4.
__device__ __forceinline__ v16h load_a_row(const _Float16* row, int hi, int k0){
  const v8h lo = *(const v8h*)(row + k0 + hi*16);
  const v8h hh = *(const v8h*)(row + k0 + hi*16 + 8);
  return __builtin_shufflevector(lo, hh, 0,1,2,3,4,5,6,7,8,9,10,11,12,13,14,15);
}

// B fragment from swizzled LDS weights: per 32x16 tile, element (k,n) lives at
// tile*512 + (16*((k&31)>>4) + (n&15))*16 + (k&15). Lane reads 16 contiguous f16.
__device__ __forceinline__ v16h load_b_sw(const _Float16* Wsw, int tile, int lane){
  const _Float16* p = Wsw + tile*512 + lane*16;
  const v8h lo = *(const v8h*)p;
  const v8h hh = *(const v8h*)(p + 8);
  return __builtin_shufflevector(lo, hh, 0,1,2,3,4,5,6,7,8,9,10,11,12,13,14,15);
}

// CDNA5 async copy: 16B per lane, global memory -> LDS, tracked by ASYNCcnt.
// ldsByteOff is the byte offset within the block's LDS allocation (dynamic LDS
// starts at 0: these kernels have no static __shared__).
__device__ __forceinline__ void async_copy16(unsigned ldsByteOff, const void* src){
  asm volatile("global_load_async_to_lds_b128 %0, %1, off"
               :: "v"(ldsByteOff), "v"(src) : "memory");
}
__device__ __forceinline__ void wait_async0(){
  asm volatile("s_wait_asynccnt 0" ::: "memory");
}

// gfx1250 hardware tanh (V_TANH_F32); fall back to libm if builtin missing.
#if __has_builtin(__builtin_amdgcn_tanhf)
__device__ __forceinline__ float htanh_(float x){ return __builtin_amdgcn_tanhf(x); }
#else
__device__ __forceinline__ float htanh_(float x){ return tanhf(x); }
#endif
__device__ __forceinline__ float sigm_(float x){ return __builtin_fmaf(0.5f, htanh_(0.5f*x), 0.5f); }

// ---------------------------------------------------------------------------
// Prep: convert + swizzle a KxN f32 weight matrix into WMMA-B fragment tiles
// (32x16 per tile, 512 f16 each), so consumer blocks do a flat LDS copy.
// ---------------------------------------------------------------------------
__global__ __launch_bounds__(256) void swizzle_kernel(
    const float* __restrict__ src, _Float16* __restrict__ dst,
    int K, int N, int NT)
{
  const int i = blockIdx.x*256 + threadIdx.x;
  if (i >= K*N) return;
  const int k = i / N, n = i % N;
  const int dl = ((k & 31) >> 4)*16 + (n & 15);
  dst[((k>>5)*NT + (n>>4))*512 + dl*16 + (k & 15)] = (_Float16)src[i];
}

// ---------------------------------------------------------------------------
// Kernel 1: LN0 -> GEMM1(128->64) -> LN1/ReLU -> concat action -> GEMM2(96->64)
//           -> LN2/ReLU -> x [BT,64] f16
// 128 threads (4 waves), 16 rows per wave, 64 rows per block.
// LDS layout: [ weights 28 tiles | sA 64x136 | sH 64x104 ]
// ---------------------------------------------------------------------------
#define SA_S 136
#define SH_S 104
#define MLP_WTILES 28           // W1: 16 tiles, W2: 12 tiles
#define MLP_SHM ((MLP_WTILES*512 + 64*SA_S + 64*SH_S)*2)

__global__ __launch_bounds__(128) void mlp_kernel(
    const float* __restrict__ ud, const float* __restrict__ act,
    const float* __restrict__ ln0_g, const float* __restrict__ ln0_b,
    const _Float16* __restrict__ wsw,          // pre-swizzled [W1|W2], 28*512 f16
    const float* __restrict__ b1,
    const float* __restrict__ ln1_g, const float* __restrict__ ln1_b,
    const float* __restrict__ b2,
    const float* __restrict__ ln2_g, const float* __restrict__ ln2_b,
    _Float16* __restrict__ xout)
{
  extern __shared__ char smem[];
  _Float16* sW1 = (_Float16*)smem;           // 16 tiles x 512
  _Float16* sW2 = sW1 + 16*512;              // 12 tiles x 512
  _Float16* sA  = sW1 + MLP_WTILES*512;      // 64 x SA_S
  _Float16* sH  = sA  + 64*SA_S;             // 64 x SH_S

  const int tid = threadIdx.x, lane = tid & 31, w = tid >> 5;
  const int hi = lane >> 4, lm = lane & 15;
  const long rowBlk = (long)blockIdx.x * 64;
  const int wrow0 = w * 16;

  // async copy of pre-swizzled weights into LDS (28*512*2 = 28672 B)
  for (int i = tid; i < MLP_WTILES*512*2/16; i += 128)
    async_copy16((unsigned)(i*16), (const char*)wsw + i*16);
  wait_async0();
  __syncthreads();

  // per-lane constant LN / bias params
  const int c0 = lane*4;
  float g0[4], bb0[4];
  #pragma unroll
  for (int j=0;j<4;++j){ g0[j]=ln0_g[c0+j]; bb0[j]=ln0_b[c0+j]; }
  float g1v[4], b1v[4], bi1[4], g2v[4], b2v[4], bi2[4];
  #pragma unroll
  for (int nt=0;nt<4;++nt){
    int n = nt*16+lm;
    g1v[nt]=ln1_g[n]; b1v[nt]=ln1_b[n]; bi1[nt]=b1[n];
    g2v[nt]=ln2_g[n]; b2v[nt]=ln2_b[n]; bi2[nt]=b2[n];
  }

  // --- LN0: one row per iteration, 32 lanes x float4 each ---
  for (int r=0;r<16;++r){
    const long grow = rowBlk + wrow0 + r;
    float4 v = *(const float4*)(ud + grow*OBS_ + c0);
    float s = v.x+v.y+v.z+v.w;
    float q = v.x*v.x+v.y*v.y+v.z*v.z+v.w*v.w;
    #pragma unroll
    for (int mk=1; mk<32; mk<<=1){ s += __shfl_xor(s,mk,32); q += __shfl_xor(q,mk,32); }
    float mean = s*(1.f/128.f);
    float var  = q*(1.f/128.f) - mean*mean;
    float rstd = rsqrtf(fmaxf(var,0.f)+1e-12f);
    v4h pk;
    pk[0]=(_Float16)((v.x-mean)*rstd*g0[0]+bb0[0]);
    pk[1]=(_Float16)((v.y-mean)*rstd*g0[1]+bb0[1]);
    pk[2]=(_Float16)((v.z-mean)*rstd*g0[2]+bb0[2]);
    pk[3]=(_Float16)((v.w-mean)*rstd*g0[3]+bb0[3]);
    *(v4h*)(sA + (wrow0+r)*SA_S + c0) = pk;
  }

  // --- GEMM1: 16 rows x 64 cols, K=128 ---
  {
    const _Float16* arow = sA + (wrow0+lm)*SA_S;
    v16h a0=load_a_row(arow,hi,0),  a1=load_a_row(arow,hi,32),
         a2=load_a_row(arow,hi,64), a3=load_a_row(arow,hi,96);
    v8f acc[4];
    #pragma unroll
    for (int nt=0;nt<4;++nt){
      v8f c = {};
      c = wmma16(a0, load_b_sw(sW1, 0*4+nt, lane), c);
      c = wmma16(a1, load_b_sw(sW1, 1*4+nt, lane), c);
      c = wmma16(a2, load_b_sw(sW1, 2*4+nt, lane), c);
      c = wmma16(a3, load_b_sw(sW1, 3*4+nt, lane), c);
      acc[nt] = c;
    }
    // bias + LN1 + ReLU -> sH cols [0,64)
    #pragma unroll
    for (int r=0;r<8;++r){
      float vv[4];
      #pragma unroll
      for (int nt=0;nt<4;++nt) vv[nt] = acc[nt][r] + bi1[nt];
      float s = vv[0]+vv[1]+vv[2]+vv[3];
      float q = vv[0]*vv[0]+vv[1]*vv[1]+vv[2]*vv[2]+vv[3]*vv[3];
      #pragma unroll
      for (int mk=1;mk<16;mk<<=1){ s += __shfl_xor(s,mk,32); q += __shfl_xor(q,mk,32); }
      float mean = s*(1.f/64.f);
      float var  = q*(1.f/64.f)-mean*mean;
      float rstd = rsqrtf(fmaxf(var,0.f)+1e-12f);
      const int mrow = wrow0 + r + 8*hi;
      #pragma unroll
      for (int nt=0;nt<4;++nt){
        float hh = (vv[nt]-mean)*rstd*g1v[nt]+b1v[nt];
        sH[mrow*SH_S + nt*16+lm] = (_Float16)fmaxf(hh,0.f);
      }
    }
  }

  // --- action -> sH cols [64,96) ---
  for (int r=0;r<16;++r){
    const long grow = rowBlk + wrow0 + r;
    sH[(wrow0+r)*SH_S + 64 + lane] = (_Float16)act[grow*ACT_ + lane];
  }

  // --- GEMM2: K=96 ---
  {
    const _Float16* arow = sH + (wrow0+lm)*SH_S;
    v16h a0=load_a_row(arow,hi,0), a1=load_a_row(arow,hi,32), a2=load_a_row(arow,hi,64);
    v8f acc[4];
    #pragma unroll
    for (int nt=0;nt<4;++nt){
      v8f c = {};
      c = wmma16(a0, load_b_sw(sW2, 0*4+nt, lane), c);
      c = wmma16(a1, load_b_sw(sW2, 1*4+nt, lane), c);
      c = wmma16(a2, load_b_sw(sW2, 2*4+nt, lane), c);
      acc[nt] = c;
    }
    // bias + LN2 + ReLU -> stage into sA (wave-private rows, same stride)
    #pragma unroll
    for (int r=0;r<8;++r){
      float vv[4];
      #pragma unroll
      for (int nt=0;nt<4;++nt) vv[nt] = acc[nt][r] + bi2[nt];
      float s = vv[0]+vv[1]+vv[2]+vv[3];
      float q = vv[0]*vv[0]+vv[1]*vv[1]+vv[2]*vv[2]+vv[3]*vv[3];
      #pragma unroll
      for (int mk=1;mk<16;mk<<=1){ s += __shfl_xor(s,mk,32); q += __shfl_xor(q,mk,32); }
      float mean = s*(1.f/64.f);
      float var  = q*(1.f/64.f)-mean*mean;
      float rstd = rsqrtf(fmaxf(var,0.f)+1e-12f);
      const int mrow = wrow0 + r + 8*hi;
      #pragma unroll
      for (int nt=0;nt<4;++nt){
        float hh = (vv[nt]-mean)*rstd*g2v[nt]+b2v[nt];
        sA[mrow*SA_S + nt*16+lm] = (_Float16)fmaxf(hh,0.f);
      }
    }
  }

  // --- coalesced copy x tile to global ---
  for (int r=0;r<16;++r){
    const long grow = rowBlk + wrow0 + r;
    ((unsigned int*)(xout + grow*H_))[lane] =
        ((const unsigned int*)(sA + (wrow0+r)*SA_S))[lane];
  }
}

// ---------------------------------------------------------------------------
// Kernel 2: LSTM (fw: bw=0 writes y[...,0:64]; bw: bw=1 writes y[...,64:128]).
// 4 waves/block, 16 batch rows per wave, h recurrence wave-private.
// Gate weights pre-swizzled: 4 K-tiles x 16 N-tiles (i|j|f|o) at LDS offset 0.
// ---------------------------------------------------------------------------
#define HT_S 72    // padded stride for 16 x 64 h / out tiles (144B, 16B-aligned)
#define LSTM_SHM ((64*512 + 64*HT_S + 64*HT_S)*2)

__global__ __launch_bounds__(128) void lstm_kernel(
    const _Float16* __restrict__ x, const _Float16* __restrict__ ksw,
    const float* __restrict__ kb, const int* __restrict__ au,
    _Float16* __restrict__ y, int bw)
{
  extern __shared__ char smem[];
  _Float16* sK  = (_Float16*)smem;           // 64 tiles x 512 (swizzled B frags)
  _Float16* sHt = sK + 64*512;               // 4 waves x 16 x HT_S
  _Float16* sOt = sHt + 64*HT_S;

  const int tid=threadIdx.x, lane=tid&31, w=tid>>5;
  const int hi=lane>>4, lm=lane&15;
  const long rowBase = (long)blockIdx.x*64 + w*16;
  const int dirOff = bw ? 64 : 0;

  // async copy of pre-swizzled gate weights into LDS (64*512*2 = 65536 B)
  for (int i=tid; i<64*512*2/16; i+=128)
    async_copy16((unsigned)(i*16), (const char*)ksw + i*16);
  wait_async0();
  __syncthreads();

  _Float16* myH = sHt + w*16*HT_S;
  _Float16* myO = sOt + w*16*HT_S;
  for (int i=lane;i<16*HT_S;i+=32) myH[i] = (_Float16)0.f;

  int au_l[8];
  #pragma unroll
  for (int r=0;r<8;++r) au_l[r] = au[rowBase + r + 8*hi];
  const int au_m = au[rowBase + lm];

  float bsg[16];
  #pragma unroll
  for (int g=0;g<4;++g){
    #pragma unroll
    for (int nt=0;nt<4;++nt) bsg[g*4+nt] = kb[g*64 + nt*16 + lm];
  }

  v8f creg[4] = {};   // cell state, C/D layout
  v8f hreg[4] = {};   // hidden state copy in registers (avoids LDS re-read)

  for (int t=0;t<T_;++t){
    // reversed-sequence time index for this lane's A-row (bw only)
    const int tx = bw ? ((t<au_m)?(au_m-1-t):t) : t;
    const _Float16* xrow = x + ((rowBase+lm)*T_ + tx)*H_;
    v16h ax0 = load_a_row(xrow,hi,0), ax1 = load_a_row(xrow,hi,32);
    const _Float16* hrow = myH + lm*HT_S;
    v16h ah0 = load_a_row(hrow,hi,0), ah1 = load_a_row(hrow,hi,32);

    #pragma unroll
    for (int nt=0;nt<4;++nt){           // hidden-column block: 16 cols
      v8f ga[4];
      #pragma unroll
      for (int g=0;g<4;++g){            // gates i, j, f, o ; N-tile = g*4+nt
        const int ntile = g*4 + nt;
        v8f c = {};
        c = wmma16(ax0, load_b_sw(sK, 0*16 + ntile, lane), c);
        c = wmma16(ax1, load_b_sw(sK, 1*16 + ntile, lane), c);
        c = wmma16(ah0, load_b_sw(sK, 2*16 + ntile, lane), c);
        c = wmma16(ah1, load_b_sw(sK, 3*16 + ntile, lane), c);
        ga[g]=c;
      }
      #pragma unroll
      for (int r=0;r<8;++r){
        float iv = ga[0][r]+bsg[0*4+nt];
        float jv = ga[1][r]+bsg[1*4+nt];
        float fv = ga[2][r]+bsg[2*4+nt];
        float ov = ga[3][r]+bsg[3*4+nt];
        float cold = creg[nt][r];
        float nc = sigm_(fv+1.0f)*cold + sigm_(iv)*htanh_(jv);
        float nh = sigm_(ov)*htanh_(nc);
        const bool valid = t < au_l[r];
        const int m = r + 8*hi, n = nt*16+lm;
        float hold = hreg[nt][r];
        float hnew = valid ? nh : hold;       // state copy-through
        creg[nt][r] = valid ? nc : cold;
        hreg[nt][r] = hnew;
        myH[m*HT_S+n] = (_Float16)hnew;                  // for next-step A-h frag
        myO[m*HT_S+n] = (_Float16)(valid ? nh : 0.0f);   // zero past length
      }
    }
    // coalesced output copy: one time slot per row (idx is an involution)
    for (int rr=0;rr<16;++rr){
      const long brow = rowBase + rr;
      int tout = t;
      if (bw){ int a = au[brow]; tout = (t<a)?(a-1-t):t; }
      ((unsigned int*)(y + (brow*T_ + tout)*128 + dirOff))[lane] =
          ((const unsigned int*)(myO + rr*HT_S))[lane];
    }
  }
}

// ---------------------------------------------------------------------------
// Kernel 3: LN over (T,2H)=640 per batch row, ReLU, conv dot, mask, sum.
// One wave per batch row.
// ---------------------------------------------------------------------------
__global__ __launch_bounds__(256) void head_kernel(
    const _Float16* __restrict__ y, const float* __restrict__ mask,
    const float* __restrict__ g3, const float* __restrict__ b3,
    const float* __restrict__ cw, const float* __restrict__ cb,
    float* __restrict__ out)
{
  const int lane = threadIdx.x & 31, w = threadIdx.x >> 5;
  const long b = (long)blockIdx.x*8 + w;
  const _Float16* yr = y + b*(T_*128);
  float vals[20];
  float s=0.f, q=0.f;
  #pragma unroll
  for (int c2=0;c2<10;++c2){             // paired loads: e = c2*64 + lane*2 (+1)
    v2h pv = *(const v2h*)(yr + c2*64 + lane*2);
    float v0 = (float)pv[0], v1 = (float)pv[1];
    vals[2*c2]=v0; vals[2*c2+1]=v1;
    s += v0+v1; q += v0*v0+v1*v1;
  }
  #pragma unroll
  for (int mk=1;mk<32;mk<<=1){ s+=__shfl_xor(s,mk,32); q+=__shfl_xor(q,mk,32); }
  const float mean = s*(1.f/640.f);
  const float var  = q*(1.f/640.f)-mean*mean;
  const float rstd = rsqrtf(fmaxf(var,0.f)+1e-12f);
  // per-lane params: col = (c2&1)*64 + lane*2 + {0,1}
  float gl[4], bl[4], wl[4];
  #pragma unroll
  for (int j=0;j<2;++j){
    #pragma unroll
    for (int p=0;p<2;++p){
      int col = j*64 + lane*2 + p;
      gl[j*2+p]=g3[col]; bl[j*2+p]=b3[col]; wl[j*2+p]=cw[col];
    }
  }
  float m5[T_];
  #pragma unroll
  for (int t=0;t<T_;++t) m5[t]=mask[b*T_+t];
  float acc=0.f;
  #pragma unroll
  for (int c2=0;c2<10;++c2){
    const int t = c2>>1, j = c2&1;
    #pragma unroll
    for (int p=0;p<2;++p){
      float v = (vals[2*c2+p]-mean)*rstd*gl[j*2+p]+bl[j*2+p];
      acc += fmaxf(v,0.f)*wl[j*2+p]*m5[t];
    }
  }
  #pragma unroll
  for (int mk=1;mk<32;mk<<=1) acc += __shfl_xor(acc,mk,32);
  if (lane==0){
    float ms = m5[0]+m5[1]+m5[2]+m5[3]+m5[4];
    out[b] = acc + cb[0]*ms;
  }
}

// ---------------------------------------------------------------------------
extern "C" void kernel_launch(void* const* d_in, const int* in_sizes, int n_in,
                              void* d_out, int out_size, void* d_ws, size_t ws_size,
                              hipStream_t stream)
{
  (void)in_sizes; (void)n_in; (void)out_size; (void)ws_size;
  const float* ud    = (const float*)d_in[0];
  const float* act   = (const float*)d_in[1];
  const float* mask  = (const float*)d_in[2];
  const float* ln0_g = (const float*)d_in[3];
  const float* ln0_b = (const float*)d_in[4];
  const float* W1    = (const float*)d_in[5];
  const float* b1    = (const float*)d_in[6];
  const float* ln1_g = (const float*)d_in[7];
  const float* ln1_b = (const float*)d_in[8];
  const float* W2    = (const float*)d_in[9];
  const float* b2    = (const float*)d_in[10];
  const float* ln2_g = (const float*)d_in[11];
  const float* ln2_b = (const float*)d_in[12];
  const float* fw_k  = (const float*)d_in[13];
  const float* fw_b  = (const float*)d_in[14];
  const float* bw_k  = (const float*)d_in[15];
  const float* bw_b  = (const float*)d_in[16];
  const float* ln3_g = (const float*)d_in[17];
  const float* ln3_b = (const float*)d_in[18];
  const float* convw = (const float*)d_in[19];
  const float* convb = (const float*)d_in[20];
  const int*   au    = (const int*)d_in[21];

  _Float16* xbuf  = (_Float16*)d_ws;                    // [BT, 64]  f16
  _Float16* ybuf  = xbuf  + (size_t)BT_*H_;             // [B, T, 128] f16 (fw|bw)
  _Float16* wmlp  = ybuf  + (size_t)BT_*128;            // 28*512 f16 (W1|W2 swizzled)
  _Float16* wfw   = wmlp  + 28*512;                     // 64*512 f16
  _Float16* wbw   = wfw   + 64*512;                     // 64*512 f16

  (void)hipFuncSetAttribute((const void*)lstm_kernel,
        hipFuncAttributeMaxDynamicSharedMemorySize, LSTM_SHM);

  // one-time weight convert + swizzle into the exact LDS image layout
  swizzle_kernel<<<(128*64 +255)/256, 256, 0, stream>>>(W1,   wmlp,          128, 64, 4);
  swizzle_kernel<<<(96*64  +255)/256, 256, 0, stream>>>(W2,   wmlp + 16*512,  96, 64, 4);
  swizzle_kernel<<<(128*256+255)/256, 256, 0, stream>>>(fw_k, wfw,           128, 256, 16);
  swizzle_kernel<<<(128*256+255)/256, 256, 0, stream>>>(bw_k, wbw,           128, 256, 16);

  mlp_kernel<<<BT_/64, 128, MLP_SHM, stream>>>(
      ud, act, ln0_g, ln0_b, wmlp, b1, ln1_g, ln1_b, b2, ln2_g, ln2_b, xbuf);
  lstm_kernel<<<B_/64, 128, LSTM_SHM, stream>>>(xbuf, wfw, fw_b, au, ybuf, 0);
  lstm_kernel<<<B_/64, 128, LSTM_SHM, stream>>>(xbuf, wbw, bw_b, au, ybuf, 1);
  head_kernel<<<B_/8, 256, 0, stream>>>(ybuf, mask, ln3_g, ln3_b, convw, convb,
                                        (float*)d_out);
}
